// GravNetOp_79534204387356
// MI455X (gfx1250) — matrix operand order repaired and checked by reading, not compiled, not run on hardware
//
#include <hip/hip_runtime.h>
#include <hip/hip_bf16.h>

typedef __attribute__((ext_vector_type(16))) _Float16     v16h;
typedef __attribute__((ext_vector_type(8)))  float        v8f;
typedef __attribute__((ext_vector_type(4)))  unsigned int u32x4;
typedef __attribute__((ext_vector_type(8)))  int          i32x8;
typedef __attribute__((ext_vector_type(4)))  int          i32x4;

#define GN_N     16384
#define GN_SEG   4
#define GN_NPER  4096
#define GN_IN    128
#define GN_SD    4
#define GN_PD    64
#define GN_OUT   128
#define GN_K     50
#define GN_FEAT  (GN_IN + 2 * GN_PD)   // 256

#if __has_builtin(__builtin_amdgcn_tensor_load_to_lds) && \
    __has_builtin(__builtin_amdgcn_s_wait_tensorcnt)
#define GN_HAVE_TDM 1
#endif

// ---------------- WMMA helpers ----------------

__device__ __forceinline__ v8f wmma_f16(v16h a, v16h b, v8f c) {
    // (neg_a, A, neg_b, B, c_mod, C, reuse_a, reuse_b)
    return __builtin_amdgcn_wmma_f32_16x16x32_f16(
        false, a, false, b, (short)0, c, false, false);
}

// A tile: 16x32 from row-major f32 [*, ld]; lane holds row M = lane%16,
// half i <-> K = (i/8)*16 + (lane<16?0:8) + i%8   (ISA 7.12.2)
__device__ __forceinline__ v16h load_a_f32(const float* __restrict__ src, int ld,
                                           int row_base, int k_base, int lane) {
    int row = row_base + (lane & 15);
    int kb  = (lane & 16) ? 8 : 0;
    const float* p = src + (size_t)row * ld + k_base + kb;
    v16h a;
#pragma unroll
    for (int g = 0; g < 2; ++g)
#pragma unroll
        for (int j = 0; j < 8; ++j)
            a[g * 8 + j] = (_Float16)p[g * 16 + j];
    return a;
}

// B tile: 32x16 from row-major f32 W [K x Nw]; lane holds col N = lane%16,
// half i <-> K = (i/8)*16 + (lane<16?0:8) + i%8
__device__ __forceinline__ v16h load_b_f32(const float* __restrict__ w, int ldb,
                                           int k_base, int col_base, int lane) {
    int col = col_base + (lane & 15);
    int kb  = (lane & 16) ? 8 : 0;
    const float* p = w + (size_t)(k_base + kb) * ldb + col;
    v16h b;
#pragma unroll
    for (int i = 0; i < 16; ++i)
        b[i] = (_Float16)p[(size_t)(i + (i & 8)) * ldb];
    return b;
}

// D tile: VGPR r -> M = r + (lane<16?0:8), N = lane%16
__device__ __forceinline__ void store_d(float* __restrict__ dst, int ldd,
                                        int row_base, int col_base, int lane, v8f c) {
    int n    = col_base + (lane & 15);
    int mofs = (lane & 16) ? 8 : 0;
#pragma unroll
    for (int r = 0; r < 8; ++r)
        dst[(size_t)(row_base + mofs + r) * ldd + n] = c[r];
}

// ---------------- Kernel 0: pack weights into f16 WMMA-B layout ----------------
// W_prop: 4 ntiles x 4 kchunks (tiles 0..15), W_out: 8 ntiles x 8 kchunks (tiles 16..79).
// Packed tile = 512 halves; lane's v16h stored contiguously (32B) -> GEMM loads b128.
__global__ void gn_pack_w_kernel(const float* __restrict__ Wp,
                                 const float* __restrict__ Wo,
                                 _Float16* __restrict__ packWp,
                                 _Float16* __restrict__ packWo) {
    int tile = blockIdx.x;            // 0..79
    int lane = threadIdx.x;           // 32 threads = 1 wave
    v16h b;
    _Float16* dst;
    if (tile < 16) {                  // W_prop tile = nt*4 + kc
        int nt = tile >> 2, kc = tile & 3;
        b   = load_b_f32(Wp, GN_PD, kc * 32, nt * 16, lane);
        dst = packWp + (size_t)tile * 512;
    } else {                          // W_out tile = nt*8 + kc
        int t = tile - 16;
        int nt = t >> 3, kc = t & 7;
        b   = load_b_f32(Wo, GN_OUT, kc * 32, nt * 16, lane);
        dst = packWo + (size_t)t * 512;
    }
    *(v16h*)(dst + (size_t)lane * 16) = b;
}

// ---------------- Kernel 1a: space = x @ W_space + b_space ----------------
__global__ void gn_space_kernel(const float* __restrict__ x,
                                const float* __restrict__ W,
                                const float* __restrict__ b,
                                float* __restrict__ spo) {
    int t = blockIdx.x * blockDim.x + threadIdx.x;   // N*SD threads
    int i = t >> 2;
    int c = t & 3;
    const float* xr = x + (size_t)i * GN_IN;
    float acc = b[c];
#pragma unroll 8
    for (int k = 0; k < GN_IN; ++k) acc += xr[k] * W[k * GN_SD + c];
    spo[t] = acc;
}

// ---------------- Kernel 1b: prop = x @ W_prop + b_prop (WMMA) ----------------
// 128 thr = 4 waves; block covers 64 rows; wave w -> cols 16w..16w+15.
// B (4 kchunks) held in registers across 4 row tiles.
__global__ void gn_prop_kernel(const float* __restrict__ x,
                               const _Float16* __restrict__ packWp,
                               const float* __restrict__ bp,
                               float* __restrict__ prop) {
    int wave = threadIdx.x >> 5;
    int lane = threadIdx.x & 31;
    int row0 = blockIdx.x * 64;
    v16h B[4];
#pragma unroll
    for (int kc = 0; kc < 4; ++kc)
        B[kc] = *(const v16h*)(packWp + (size_t)(wave * 4 + kc) * 512 + (size_t)lane * 16);
    float bias = bp[wave * 16 + (lane & 15)];
#pragma unroll
    for (int rt = 0; rt < 4; ++rt) {
        int row_base = row0 + rt * 16;
        v8f c = {};
#pragma unroll
        for (int kc = 0; kc < 4; ++kc) {
            v16h a = load_a_f32(x, GN_IN, row_base, kc * 32, lane);
            c = wmma_f16(a, B[kc], c);
        }
#pragma unroll
        for (int r = 0; r < 8; ++r) c[r] += bias;
        store_d(prop, GN_PD, row_base, wave * 16, lane, c);
    }
}

// ---------------- Kernel 2: per-segment brute-force KNN (K=50) ----------------
// Segment coords (64 KB, float4) staged into LDS via TDM; per-thread sorted
// top-K lists in LDS (50 KB). One thread per query.
__global__ void gn_knn_kernel(const float* __restrict__ space,
                              int* __restrict__ nbr, float* __restrict__ d2o) {
    __shared__ float4 sc[GN_NPER];               // 64 KB
    __shared__ float  sdist[128 * GN_K];         // 25.6 KB
    __shared__ int    sidx[128 * GN_K];          // 25.6 KB

    int seg  = blockIdx.y;
    int base = seg * GN_NPER;

#ifdef GN_HAVE_TDM
    if ((threadIdx.x >> 5) == 0) {
        // Tensor DMA descriptor (CDNA5 ISA):
        // 1 descriptor, 4-byte elements, tile 16384x1, tensor 16384x1.
        unsigned long long ga = (unsigned long long)(const void*)(space + (size_t)base * GN_SD);
        unsigned int       la = (unsigned int)(size_t)(void*)&sc[0];
        u32x4 g0;
        g0.x = 1u;                                             // count=1, user mode
        g0.y = la;                                             // lds_addr
        g0.z = (unsigned int)ga;                               // global_addr[31:0]
        g0.w = (unsigned int)((ga >> 32) & 0x01FFFFFFu) | (2u << 30);  // [56:32] + type=2
        i32x8 g1;
        g1[0] = 0x00020000;   // workgroup_mask=0, data_size=2 (4B)
        g1[1] = 0x40000000;   // tensor_dim0 = 16384 (low 16 bits in [31:16])
        g1[2] = 0x00010000;   // tensor_dim0 hi=0; tensor_dim1 = 1 (low bits)
        g1[3] = 0x40000000;   // tensor_dim1 hi=0; tile_dim0 = 16384
        g1[4] = 0x00000001;   // tile_dim1 = 1; tile_dim2 = 0
        g1[5] = 16384;        // tensor_dim0_stride (low 32)
        g1[6] = 0;
        g1[7] = 0;
        i32x4 gz = {0, 0, 0, 0};
#if __clang_major__ >= 23
        i32x8 gz8 = {0, 0, 0, 0, 0, 0, 0, 0};
        __builtin_amdgcn_tensor_load_to_lds(g0, g1, gz, gz, gz8, 0);
#else
        __builtin_amdgcn_tensor_load_to_lds(g0, g1, gz, gz, 0);
#endif
        __builtin_amdgcn_s_wait_tensorcnt(0);
    }
    __syncthreads();
#else
    {
        const float4* g = (const float4*)(space + (size_t)base * GN_SD);
        for (int t = threadIdx.x; t < GN_NPER; t += blockDim.x) sc[t] = g[t];
        __syncthreads();
    }
#endif

    int q = blockIdx.x * blockDim.x + threadIdx.x;   // local query index in segment
    float4 qc = sc[q];

    float* dl = sdist + threadIdx.x * GN_K;
    int*   il = sidx  + threadIdx.x * GN_K;
    int   cnt   = 0;
    float worst = 3.4e38f;

#pragma unroll 4
    for (int j = 0; j < GN_NPER; ++j) {
        float4 cj = sc[j];                           // one ds_load_b128
        float a = cj.x - qc.x, b = cj.y - qc.y;
        float c = cj.z - qc.z, e = cj.w - qc.w;
        float d2v = a * a + b * b + c * c + e * e;
        if (d2v < worst) {                           // worst==INF until list full
            int p = (cnt < GN_K) ? cnt : (GN_K - 1);
            while (p > 0 && dl[p - 1] > d2v) {       // strict '>' : lowest-index ties kept
                dl[p] = dl[p - 1]; il[p] = il[p - 1]; --p;
            }
            dl[p] = d2v; il[p] = j;
            if (cnt < GN_K) ++cnt;
            if (cnt == GN_K) worst = dl[GN_K - 1];
        }
    }

    int*   no = nbr + (size_t)(base + q) * GN_K;
    float* dn = d2o + (size_t)(base + q) * GN_K;
    for (int k = 0; k < GN_K; ++k) { no[k] = base + il[k]; dn[k] = dl[k]; }
}

// ---------------- Kernel 3: gather/aggregate + assemble feat ----------------
// 256 thr = 4 nodes x 64 feature dims
__global__ void gn_agg_kernel(const float* __restrict__ x,
                              const float* __restrict__ prop,
                              const int* __restrict__ nbr,
                              const float* __restrict__ d2,
                              float* __restrict__ feat) {
    int qn = threadIdx.x >> 6;
    int d  = threadIdx.x & 63;
    int i  = blockIdx.x * 4 + qn;
    const int*   nl = nbr + (size_t)i * GN_K;
    const float* dl = d2  + (size_t)i * GN_K;
    float mean = 0.f, mx = -3.4e38f;
#pragma unroll 5
    for (int k = 0; k < GN_K; ++k) {
        int   j = nl[k];
        float w = __expf(-10.f * dl[k]);
        float v = prop[(size_t)j * GN_PD + d] * w;
        mean += v;
        mx = fmaxf(mx, v);
    }
    float* fr = feat + (size_t)i * GN_FEAT;
    fr[GN_IN + d]         = mean * (1.0f / (float)GN_K);
    fr[GN_IN + GN_PD + d] = mx;
    const float* xr = x + (size_t)i * GN_IN;
    fr[d]      = xr[d];
    fr[d + 64] = xr[d + 64];
}

// ---------------- Kernel 4: out = relu(feat @ W_out + b_out) (WMMA) ----------------
// 256 thr = 8 waves; block covers 64 rows; wave w -> cols 16w..16w+15.
// B (8 kchunks, 64 VGPRs) held across 4 row tiles.
__global__ void gn_out_kernel(const float* __restrict__ feat,
                              const _Float16* __restrict__ packWo,
                              const float* __restrict__ bo,
                              float* __restrict__ out) {
    int wave = threadIdx.x >> 5;
    int lane = threadIdx.x & 31;
    int row0 = blockIdx.x * 64;
    v16h B[8];
#pragma unroll
    for (int kc = 0; kc < 8; ++kc)
        B[kc] = *(const v16h*)(packWo + (size_t)(wave * 8 + kc) * 512 + (size_t)lane * 16);
    float bias = bo[wave * 16 + (lane & 15)];
#pragma unroll
    for (int rt = 0; rt < 4; ++rt) {
        int row_base = row0 + rt * 16;
        v8f c = {};
#pragma unroll
        for (int kc = 0; kc < 8; ++kc) {
            v16h a = load_a_f32(feat, GN_FEAT, row_base, kc * 32, lane);
            c = wmma_f16(a, B[kc], c);
        }
#pragma unroll
        for (int r = 0; r < 8; ++r) c[r] = fmaxf(c[r] + bias, 0.f);
        store_d(out, GN_OUT, row_base, wave * 16, lane, c);
    }
}

// ---------------- launch ----------------
extern "C" void kernel_launch(void* const* d_in, const int* in_sizes, int n_in,
                              void* d_out, int out_size, void* d_ws, size_t ws_size,
                              hipStream_t stream) {
    const float* x   = (const float*)d_in[0];
    // d_in[1] row_splits: fixed equal segments, layout hardcoded
    const float* Wsp = (const float*)d_in[2];
    const float* bsp = (const float*)d_in[3];
    const float* Wpr = (const float*)d_in[4];
    const float* bpr = (const float*)d_in[5];
    const float* Wo  = (const float*)d_in[6];
    const float* bo  = (const float*)d_in[7];

    // output tuple, flat in return order: out, neighbor_idx, distsq, space
    float* out = (float*)d_out;                              // N*128 f32
    int*   nbr = (int*)(out + (size_t)GN_N * GN_OUT);        // N*50 i32
    float* d2o = (float*)(nbr + (size_t)GN_N * GN_K);        // N*50 f32
    float* spo = d2o + (size_t)GN_N * GN_K;                  // N*4  f32

    // workspace (all regions 32B-aligned)
    float*    prop   = (float*)d_ws;                               // N*64  f32 (4 MB)
    float*    feat   = prop + (size_t)GN_N * GN_PD;                // N*256 f32 (16 MB)
    _Float16* packWp = (_Float16*)(feat + (size_t)GN_N * GN_FEAT); // 16*512 f16 (16 KB)
    _Float16* packWo = packWp + 16 * 512;                          // 64*512 f16 (64 KB)

    gn_pack_w_kernel<<<80, 32, 0, stream>>>(Wpr, Wo, packWp, packWo);
    gn_space_kernel <<<(GN_N * GN_SD) / 256, 256, 0, stream>>>(x, Wsp, bsp, spo);
    gn_prop_kernel  <<<GN_N / 64, 128, 0, stream>>>(x, packWp, bpr, prop);
    gn_knn_kernel   <<<dim3(GN_NPER / 128, GN_SEG), 128, 0, stream>>>(spo, nbr, d2o);
    gn_agg_kernel   <<<GN_N / 4, 256, 0, stream>>>(x, prop, nbr, d2o, feat);
    gn_out_kernel   <<<GN_N / 64, 256, 0, stream>>>(feat, packWo, bo, out);
}
